// DFMConv2d_34892314312792
// MI455X (gfx1250) — matrix operand order repaired and checked by reading, not compiled, not run on hardware
//
#include <hip/hip_runtime.h>
#include <cstdint>

typedef __attribute__((ext_vector_type(16))) _Float16 v16h;
typedef __attribute__((ext_vector_type(8)))  float    v8f;

#define C_IN  256
#define C_OUT 256
#define NB    8
#define HID   128
#define HW    64
#define NBATCH 64

// workspace layout (bytes)
#define WS_POOLED 0                       // 64*256 f32      = 65536
#define WS_MIXH   65536                   // 64*256*8 f16    = 262144
#define WS_BPACK  (65536 + 262144)        // 9*8*32*8 u32    = 73728

union AFrag { v16h h; uint32_t u[8]; };
union BFrag { v16h h; uint32_t u[8]; uint4 q[2]; };

// ---------------------------------------------------------------------------
// Kernel 0: repack base filters into WMMA B-fragment layout, f16.
// Index: ((t*8+cc)*32 + lane)*8 + v  ->  B[k,n] pair, k=(lane<16?0:16)+2v,
// n=lane%16 (n>=8 -> zero column), channel c = cc*32+k, tap t = dh*3+dw.
// ---------------------------------------------------------------------------
__global__ void pack_filters(const float* __restrict__ bfil,
                             uint32_t* __restrict__ bp) {
    int g = blockIdx.x * blockDim.x + threadIdx.x;
    if (g >= 9 * 8 * 32 * 8) return;
    int v  = g & 7;
    int l  = (g >> 3) & 31;
    int cc = (g >> 8) & 7;
    int t  = g >> 11;
    int n  = l & 15;
    int k  = ((l < 16) ? 0 : 16) + v * 2;
    int c  = cc * 32 + k;
    float f0 = 0.f, f1 = 0.f;
    if (n < NB) {
        f0 = bfil[n * 2304 + c * 9 + t];
        f1 = bfil[n * 2304 + (c + 1) * 9 + t];
    }
    union { _Float16 h[2]; uint32_t u; } p;
    p.h[0] = (_Float16)f0;
    p.h[1] = (_Float16)f1;
    bp[g] = p.u;
}

// ---------------------------------------------------------------------------
// Kernel 1: global average pool. One block per (channel, sample).
// ---------------------------------------------------------------------------
__global__ void pool_kernel(const float* __restrict__ x,
                            float* __restrict__ pooled) {
    int c = blockIdx.x, b = blockIdx.y;
    const float* p = x + (size_t)(b * C_IN + c) * (HW * HW);
    float s = 0.f;
    for (int i = threadIdx.x; i < HW * HW; i += 256) s += p[i];
    __shared__ float red[256];
    red[threadIdx.x] = s;
    __syncthreads();
    for (int off = 128; off > 0; off >>= 1) {
        if (threadIdx.x < off) red[threadIdx.x] += red[threadIdx.x + off];
        __syncthreads();
    }
    if (threadIdx.x == 0) pooled[b * C_IN + c] = red[0] * (1.0f / (HW * HW));
}

// ---------------------------------------------------------------------------
// Kernel 2: MLP + softmax -> mix weights as f16. One block per sample.
// ---------------------------------------------------------------------------
__global__ void mlp_kernel(const float* __restrict__ pooled,
                           const float* __restrict__ w1,
                           const float* __restrict__ b1,
                           const float* __restrict__ w2,
                           const float* __restrict__ b2,
                           _Float16* __restrict__ mixh) {
    int b   = blockIdx.x;
    int tid = threadIdx.x;
    __shared__ float ps[C_IN];
    __shared__ float hs[HID];
    ps[tid] = pooled[b * C_IN + tid];
    __syncthreads();
    if (tid < HID) {
        float acc = b1[tid];
        const float* wr = w1 + tid * C_IN;
        for (int i = 0; i < C_IN; ++i) acc += ps[i] * wr[i];
        hs[tid] = acc > 0.f ? acc : 0.f;
    }
    __syncthreads();
    int o = tid;
    float lg[NB];
    for (int n = 0; n < NB; ++n) {
        const float* wr = w2 + (o * NB + n) * HID;
        float acc = b2[o * NB + n];
        for (int i = 0; i < HID; ++i) acc += hs[i] * wr[i];
        lg[n] = acc;
    }
    float m = lg[0];
    for (int n = 1; n < NB; ++n) m = fmaxf(m, lg[n]);
    float s = 0.f;
    for (int n = 0; n < NB; ++n) { lg[n] = __expf(lg[n] - m); s += lg[n]; }
    float inv = 1.0f / s;
    for (int n = 0; n < NB; ++n)
        mixh[(size_t)(b * C_OUT + o) * NB + n] = (_Float16)(lg[n] * inv);
}

// ---------------------------------------------------------------------------
// Kernel 3: fused tap-major implicit-GEMM conv (WMMA f16) + mixing GEMM (WMMA).
// One block = one sample x one 16x16 spatial tile. 8 waves; wave w owns
// spatial rows {2w, 2w+1} for the conv and o-tiles {2w, 2w+1} for the mix.
// ---------------------------------------------------------------------------
__global__ __launch_bounds__(256) void conv_mix_kernel(
    const float* __restrict__ x, const _Float16* __restrict__ mixh,
    const uint32_t* __restrict__ bp, float* __restrict__ out) {
    int b    = blockIdx.y;
    int tx   = blockIdx.x & 3;
    int ty   = blockIdx.x >> 2;
    int tid  = threadIdx.x;
    int lane = tid & 31;
    int wave = tid >> 5;
    int l16  = (lane < 16) ? 0 : 1;
    int px   = lane & 15;
    int py0  = wave * 2;
    int py1  = wave * 2 + 1;

    // x tile: 18x18 halo cells, 32 channels/cell, stride 34 halves (17 banks)
    __shared__ __align__(16) _Float16 xs[18 * 18 * 34];
    // y tile: 256 positions, K padded 0..31 (+2 pad), stride 34 halves
    __shared__ __align__(16) _Float16 ys[256 * 34];

    for (int i = tid; i < 256 * 34; i += 256) ys[i] = (_Float16)0.f;

    v8f acc0 = {};
    v8f acc1 = {};
    const size_t xbase = (size_t)b * C_IN * HW * HW;

    for (int cc = 0; cc < 8; ++cc) {
        __syncthreads();                       // xs free (and ys zero done)
        // stage 18x18x32 f32->f16 tile into LDS
        for (int i = tid; i < 32 * 18 * 18; i += 256) {
            int co  = i / 324;
            int rem = i - co * 324;
            int hh  = rem / 18;
            int wwi = rem - hh * 18;
            int gh  = ty * 16 + hh - 1;
            int gw  = tx * 16 + wwi - 1;
            float fv = 0.f;
            if (gh >= 0 && gh < HW && gw >= 0 && gw < HW)
                fv = x[xbase + (size_t)(cc * 32 + co) * HW * HW + gh * HW + gw];
            xs[(hh * 18 + wwi) * 34 + co] = (_Float16)fv;
        }
        __syncthreads();
        for (int t = 0; t < 9; ++t) {
            int dh = t / 3, dw = t - dh * 3;
            BFrag bm;
            const uint4* bq = (const uint4*)bp + (size_t)((t * 8 + cc) * 32 + lane) * 2;
            bm.q[0] = bq[0];
            bm.q[1] = bq[1];

            AFrag a;
            int cellw = px + dw;
            int cell0 = (py0 + dh) * 18 + cellw;
#pragma unroll
            for (int v = 0; v < 8; ++v) {
                int k = l16 * 16 + v * 2;   // A k-map
                a.u[v] = *(const uint32_t*)&xs[cell0 * 34 + k];
            }
            acc0 = __builtin_amdgcn_wmma_f32_16x16x32_f16(
                false, a.h, false, bm.h, (short)0, acc0, false, false);

            int cell1 = (py1 + dh) * 18 + cellw;
#pragma unroll
            for (int v = 0; v < 8; ++v) {
                int k = l16 * 16 + v * 2;
                a.u[v] = *(const uint32_t*)&xs[cell1 * 34 + k];
            }
            acc1 = __builtin_amdgcn_wmma_f32_16x16x32_f16(
                false, a.h, false, bm.h, (short)0, acc1, false, false);
        }
    }

    // y -> LDS as f16 (cols 8..15 are exact zeros from zero B columns;
    // cols 16..31 remain zero from the init)
#pragma unroll
    for (int r = 0; r < 8; ++r) {
        int pxr = r + l16 * 8;
        int n   = lane & 15;
        ys[(py0 * 16 + pxr) * 34 + n] = (_Float16)acc0[r];
        ys[(py1 * 16 + pxr) * 34 + n] = (_Float16)acc1[r];
    }
    __syncthreads();

    // mixing GEMM: out[o,p] = mix[o, 0..7] @ y[0..7, p], K padded to 32
    const uint32_t* mixw = (const uint32_t*)mixh;
#pragma unroll
    for (int oi = 0; oi < 2; ++oi) {
        int ot = wave * 2 + oi;
        AFrag am;
#pragma unroll
        for (int v = 0; v < 8; ++v) {
            uint32_t val = 0;
            if (lane < 16 && v < 4)            // k pairs (2v,2v+1), k<8 real
                val = mixw[(size_t)(b * C_OUT + ot * 16 + lane) * 4 + v];
            am.u[v] = val;
        }
        for (int pt = 0; pt < 16; ++pt) {
            BFrag bm;
#pragma unroll
            for (int v = 0; v < 8; ++v) {
                int k = l16 * 16 + v * 2;       // B k-map
                bm.u[v] = *(const uint32_t*)&ys[(pt * 16 + (lane & 15)) * 34 + k];
            }
            v8f d = {};
            d = __builtin_amdgcn_wmma_f32_16x16x32_f16(
                false, am.h, false, bm.h, (short)0, d, false, false);
#pragma unroll
            for (int r = 0; r < 8; ++r) {
                int o  = ot * 16 + r + l16 * 8;
                int gh = ty * 16 + pt;
                int gw = tx * 16 + (lane & 15);
                out[(((size_t)b * C_OUT + o) * HW + gh) * HW + gw] = d[r];
            }
        }
    }
}

// ---------------------------------------------------------------------------
extern "C" void kernel_launch(void* const* d_in, const int* in_sizes, int n_in,
                              void* d_out, int out_size, void* d_ws, size_t ws_size,
                              hipStream_t stream) {
    const float* x    = (const float*)d_in[0];
    const float* w1   = (const float*)d_in[1];
    const float* b1   = (const float*)d_in[2];
    const float* w2   = (const float*)d_in[3];
    const float* b2   = (const float*)d_in[4];
    const float* bfil = (const float*)d_in[5];
    float* out = (float*)d_out;

    char* ws = (char*)d_ws;
    float*    pooled = (float*)(ws + WS_POOLED);
    _Float16* mixh   = (_Float16*)(ws + WS_MIXH);
    uint32_t* bp     = (uint32_t*)(ws + WS_BPACK);

    pack_filters<<<72, 256, 0, stream>>>(bfil, bp);
    pool_kernel<<<dim3(C_IN, NBATCH), 256, 0, stream>>>(x, pooled);
    mlp_kernel<<<NBATCH, 256, 0, stream>>>(pooled, w1, b1, w2, b2, mixh);
    conv_mix_kernel<<<dim3(16, NBATCH), 256, 0, stream>>>(x, mixh, bp, out);
}
